// RNNLM_23252952940827
// MI455X (gfx1250) — compile-verified, matrix operand order checked by reading, not run on hardware
//
#include <hip/hip_runtime.h>
#include <stddef.h>
#include <stdint.h>

// ---------------- problem constants ----------------
#define V_   8192
#define H_   512
#define E_   256
#define B_   32
#define T_   1024
#define BT_  (B_*T_)      // 32768
#define G3H  (3*H_)       // 1536
#define NWG_GRU 8         // GRU workgroups: 8 x 64 columns of h

// d_out layout (f32): logits [B,T,V], carry [B,H], emb [B,T,E]
static const size_t OUT_CARRY = (size_t)BT_ * V_;
static const size_t OUT_EMB   = OUT_CARRY + (size_t)B_ * H_;

// ---------------- workspace layout (bytes, 256-aligned) ----------------
static const size_t WS_HBUF  = 0;                                    // 2*B*H f32 (h ping-pong)
static const size_t WS_HBF   = WS_HBUF + 2ull * B_ * H_ * 4;          // 2*B*H u16 (bf16 mirror)
static const size_t WS_BAR   = WS_HBF + 2ull * B_ * H_ * 2;           // 2 ints (count, gen)
static const size_t WS_BIAS  = WS_BAR + 256;                          // 1536 f32
static const size_t WS_WI_PK = WS_BIAS + (size_t)G3H * 4;             // [1536][E/2] u32 (bf16x2)
static const size_t WS_WH_PK = WS_WI_PK + (size_t)(E_/2) * G3H * 4;   // [1536][H/2] u32
static const size_t WS_WD_PK = WS_WH_PK + (size_t)(H_/2) * G3H * 4;   // [V][H/2] u32
static const size_t WS_EMBA  = WS_WD_PK + (size_t)(H_/2) * V_ * 4;    // [T*B, E] bf16 (u16)
static const size_t WS_HS    = WS_EMBA + (size_t)BT_ * E_ * 2;        // [B*T, H] bf16 (u16)
static const size_t WS_GI    = WS_HS + (size_t)BT_ * H_ * 2;          // [T*B, 3H] f32
// total ~= 262.6 MB

// ---------------- CDNA5 feature detection (compile-probe, never fails the build) ----------
#ifndef __has_builtin
#define __has_builtin(x) 0
#endif
#if __has_builtin(__builtin_amdgcn_global_load_async_to_lds_b32) && \
    __has_builtin(__builtin_amdgcn_global_load_async_to_lds_b128) && \
    __has_builtin(__builtin_amdgcn_s_wait_asynccnt)
#define HAVE_ASYNC 1
#else
#define HAVE_ASYNC 0
#endif
#if __has_builtin(__builtin_amdgcn_tensor_load_to_lds) && \
    __has_builtin(__builtin_amdgcn_s_wait_tensorcnt)
#define HAVE_TDM 1
#else
#define HAVE_TDM 0
#endif

typedef int vi4 __attribute__((vector_size(16)));                 // param pointee for b128
typedef __attribute__((address_space(1))) int  gas_i32;           // AS1 ("__device__") int
typedef __attribute__((address_space(3))) int  las_i32;           // AS3 ("__shared__") int
typedef __attribute__((address_space(1))) vi4  gas_vi4;
typedef __attribute__((address_space(3))) vi4  las_vi4;
typedef unsigned int u32x4 __attribute__((ext_vector_type(4)));   // trivial 16B quad
typedef int          i32x4 __attribute__((ext_vector_type(4)));
typedef int          i32x8 __attribute__((ext_vector_type(8)));

#if HAVE_ASYNC
__device__ __forceinline__ void async_b32(const void* g, void* l) {
    __builtin_amdgcn_global_load_async_to_lds_b32(
        (gas_i32*)(uintptr_t)g, (las_i32*)(unsigned int)(uintptr_t)l, 0, 0);
}
__device__ __forceinline__ void async_b128(const void* g, void* l) {
    __builtin_amdgcn_global_load_async_to_lds_b128(
        (gas_vi4*)(uintptr_t)g, (las_vi4*)(unsigned int)(uintptr_t)l, 0, 0);
}
#endif

// ---------------- types for WMMA ----------------
typedef __bf16 bf16_t;
typedef bf16_t v16bf __attribute__((ext_vector_type(16)));
typedef float  v8f   __attribute__((ext_vector_type(8)));

// Fragment: u[0..3] = k-pairs {4h..4h+3}, u[4..7] = k-pairs {8+4h..8+4h+3} at one column.
// With LDS laid out [col][kpair], each half is one 16B contiguous quad -> ds_load_b128.
union Frag { unsigned int u[8]; u32x4 q[2]; v16bf v; };

__device__ __forceinline__ unsigned short f32_to_bf16_bits(float f) {
    unsigned int u = __builtin_bit_cast(unsigned int, f);
    unsigned int r = u + 0x7FFFu + ((u >> 16) & 1u);   // round-to-nearest-even
    return (unsigned short)(r >> 16);
}
__device__ __forceinline__ unsigned int pack_bf16x2(float lo, float hi) {
    return (unsigned int)f32_to_bf16_bits(lo) | ((unsigned int)f32_to_bf16_bits(hi) << 16);
}
__device__ __forceinline__ float sigmoidf_(float x) {
    return 1.0f / (1.0f + __expf(-x));
}

// ---------------- init / pack kernels ----------------
__global__ __launch_bounds__(256) void init_kernel(float* hbuf, unsigned int* hbf32,
                                                   int* bar,
                                                   const float* bi_r, const float* bi_z,
                                                   const float* bi_n, float* bias_cat) {
    int idx = blockIdx.x * 256 + threadIdx.x;
    if (idx < 2 * B_ * H_) hbuf[idx] = 0.0f;
    if (idx < B_ * H_)     hbf32[idx] = 0u;      // 2*B*H u16 = B*H u32 of bf16 zeros
    if (idx < 2) bar[idx] = 0;
    if (idx < H_)            bias_cat[idx] = bi_r[idx];
    else if (idx < 2*H_)     bias_cat[idx] = bi_z[idx - H_];
    else if (idx < 3*H_)     bias_cat[idx] = bi_n[idx - 2*H_];
}

// B pack, k-major per column: dst[n*(K/2) + kp] = bf16x2(src_gate[2kp][n%512], src_gate[2kp+1][n%512])
__global__ __launch_bounds__(256) void pack3_kernel(const float* __restrict__ s0,
                                                    const float* __restrict__ s1,
                                                    const float* __restrict__ s2,
                                                    unsigned int* __restrict__ dst, int K) {
    int idx = blockIdx.x * 256 + threadIdx.x;
    if (idx >= (K / 2) * G3H) return;
    int kp = idx % (K / 2);
    int n  = idx / (K / 2);
    const float* s = (n < H_) ? s0 : ((n < 2*H_) ? s1 : s2);
    int nn = n & (H_ - 1);
    dst[idx] = pack_bf16x2(s[(2*kp) * H_ + nn], s[(2*kp + 1) * H_ + nn]);
}

// dst[n*(K/2) + kp] = bf16x2(src[2kp][n], src[2kp+1][n])
__global__ __launch_bounds__(256) void pack1_kernel(const float* __restrict__ src,
                                                    unsigned int* __restrict__ dst,
                                                    int K, int N) {
    int idx = blockIdx.x * 256 + threadIdx.x;
    if (idx >= (K / 2) * N) return;
    int kp = idx % (K / 2);
    int n  = idx / (K / 2);
    dst[idx] = pack_bf16x2(src[(size_t)(2*kp) * N + n], src[(size_t)(2*kp + 1) * N + n]);
}

// emb gather: d_out emb section f32 [B,T,E]; bf16 A-matrix [T*B, E] for the input GEMM
__global__ __launch_bounds__(256) void embed_kernel(const int* __restrict__ x,
                                                    const float* __restrict__ embed,
                                                    float* __restrict__ emb_out,
                                                    unsigned short* __restrict__ embA) {
    size_t idx = (size_t)blockIdx.x * 256 + threadIdx.x;   // b*T*E + t*E + e
    int e = (int)(idx & (E_ - 1));
    size_t bt = idx >> 8;                                   // b*T + t
    int t = (int)(bt & (T_ - 1));
    int b = (int)(bt >> 10);
    int tok = x[bt];
    float f = embed[(size_t)tok * E_ + e];
    emb_out[idx] = f;
    embA[((size_t)t * B_ + b) * E_ + e] = f32_to_bf16_bits(f);
}

// ---------------- generic bf16 WMMA GEMM ----------------
// C[M,N] f32 = A[M,K]bf16 @ B + bias.  A row-major u16; Bq k-pair packed u32 [N][K/2].
// Block 128x128, 256 thr = 8 waves (2x4), wave tile 64x32 = 4x2 WMMA C-tiles.
// LDS layout [col][kpair16] so fragments are two ds_load_b128 each.
// Double-buffered LDS; staging = 4 async_b128/thread per 32-deep k slab.
#define GEMM_STAGE_OPS 4

__device__ __forceinline__ void gemm_stage(const unsigned short* __restrict__ A,
                                           const unsigned int* __restrict__ Bq,
                                           int N, int K, int blockM, int blockN,
                                           int k0, int tid,
                                           unsigned int* la, unsigned int* lb) {
#if HAVE_ASYNC
    #pragma unroll
    for (int it = 0; it < 2; ++it) {            // A: 128 rows x 4 quads (k-contiguous rows)
        int idx = tid + it * 256;
        int m = idx >> 2, qd = idx & 3;
        async_b128(A + (blockM + m) * K + k0 + qd * 8, &la[m * 16 + qd * 4]);
    }
    #pragma unroll
    for (int it = 0; it < 2; ++it) {            // B: 128 cols x 4 quads (k-major pack)
        int idx = tid + it * 256;
        int n = idx >> 2, qd = idx & 3;
        async_b128(Bq + (blockN + n) * (K >> 1) + (k0 >> 1) + qd * 4,
                   &lb[n * 16 + qd * 4]);
    }
#else
    #pragma unroll
    for (int it = 0; it < 2; ++it) {
        int idx = tid + it * 256;
        int m = idx >> 2, qd = idx & 3;
        *(u32x4*)&la[m * 16 + qd * 4] =
            *(const u32x4*)(A + (blockM + m) * K + k0 + qd * 8);
    }
    #pragma unroll
    for (int it = 0; it < 2; ++it) {
        int idx = tid + it * 256;
        int n = idx >> 2, qd = idx & 3;
        *(u32x4*)&lb[n * 16 + qd * 4] =
            *(const u32x4*)(Bq + (blockN + n) * (K >> 1) + (k0 >> 1) + qd * 4);
    }
#endif
}

__global__ __launch_bounds__(256) void gemm_kernel(const unsigned short* __restrict__ A,
                                                   const unsigned int* __restrict__ Bq,
                                                   const float* __restrict__ bias,
                                                   float* __restrict__ C,
                                                   int N, int K) {
    __shared__ __align__(16) unsigned int lds_a[2][128 * 16];   // [buf][m][kpair]
    __shared__ __align__(16) unsigned int lds_b[2][128 * 16];   // [buf][n][kpair]
    const int tid  = threadIdx.x;
    const int lane = tid & 31;
    const int wave = tid >> 5;
    const int wm = wave >> 2;          // 0..1
    const int wn = wave & 3;           // 0..3
    const int blockM = blockIdx.y * 128;
    const int blockN = blockIdx.x * 128;
    const int nl   = lane & 15;
    const int half = lane >> 4;

    v8f acc[4][2] = {};

    const int S = K / 32;
    gemm_stage(A, Bq, N, K, blockM, blockN, 0, tid, lds_a[0], lds_b[0]);

    for (int s = 0; s < S; ++s) {
        const int cur = s & 1;
        if (s + 1 < S) {
            // buffer (1-cur) was last read in stage s-1; trailing barrier made that safe
            gemm_stage(A, Bq, N, K, blockM, blockN, (s + 1) * 32, tid,
                       lds_a[1 - cur], lds_b[1 - cur]);
            __builtin_prefetch(A + (blockM + (tid >> 1)) * K + (s + 1) * 32, 0, 1);
        }
#if HAVE_ASYNC
        // ASYNCcnt decrements in order: <=GEMM_STAGE_OPS means stage s is complete
        if (s + 1 < S) __builtin_amdgcn_s_wait_asynccnt(GEMM_STAGE_OPS);
        else           __builtin_amdgcn_s_wait_asynccnt(0);
#endif
        __syncthreads();

        Frag afr[4], bfr[2];
        #pragma unroll
        for (int mt = 0; mt < 4; ++mt) {
            int m = wm * 64 + mt * 16 + nl;
            afr[mt].q[0] = *(const u32x4*)&lds_a[cur][m * 16 + half * 4];
            afr[mt].q[1] = *(const u32x4*)&lds_a[cur][m * 16 + 8 + half * 4];
        }
        #pragma unroll
        for (int nt = 0; nt < 2; ++nt) {
            int n = wn * 32 + nt * 16 + nl;
            bfr[nt].q[0] = *(const u32x4*)&lds_b[cur][n * 16 + half * 4];
            bfr[nt].q[1] = *(const u32x4*)&lds_b[cur][n * 16 + 8 + half * 4];
        }
        #pragma unroll
        for (int mt = 0; mt < 4; ++mt)
            #pragma unroll
            for (int nt = 0; nt < 2; ++nt)
                acc[mt][nt] = __builtin_amdgcn_wmma_f32_16x16x32_bf16(
                    false, afr[mt].v, false, bfr[nt].v,
                    (short)0, acc[mt][nt], false, false);
        __syncthreads();   // reads of buf[cur] done before stage s+2 overwrites it
    }

    #pragma unroll
    for (int mt = 0; mt < 4; ++mt) {
        int row0 = blockM + wm * 64 + mt * 16;
        #pragma unroll
        for (int nt = 0; nt < 2; ++nt) {
            int col = blockN + wn * 32 + nt * 16 + nl;
            float bv = bias[col];
            #pragma unroll
            for (int r = 0; r < 8; ++r) {
                int row = row0 + r + half * 8;     // C/D layout: M = vgpr + 8*half
                C[(size_t)row * N + col] = acc[mt][nt][r] + bv;
            }
        }
    }
}

// ---------------- device-scope barrier for the recurrence ----------------
__device__ __forceinline__ void grid_barrier(int* cnt, int* gen, int nwg) {
    __threadfence();                          // make h_new visible at device scope
    __builtin_amdgcn_s_cluster_barrier();     // HW cluster path (NOP if not cluster-launched)
    __syncthreads();
    if (threadIdx.x == 0) {
        int g = __hip_atomic_load(gen, __ATOMIC_ACQUIRE, __HIP_MEMORY_SCOPE_AGENT);
        if (__hip_atomic_fetch_add(cnt, 1, __ATOMIC_ACQ_REL, __HIP_MEMORY_SCOPE_AGENT) == nwg - 1) {
            __hip_atomic_store(cnt, 0, __ATOMIC_RELAXED, __HIP_MEMORY_SCOPE_AGENT);
            __hip_atomic_fetch_add(gen, 1, __ATOMIC_ACQ_REL, __HIP_MEMORY_SCOPE_AGENT);
        } else {
            while (__hip_atomic_load(gen, __ATOMIC_ACQUIRE, __HIP_MEMORY_SCOPE_AGENT) == g)
                __builtin_amdgcn_s_sleep(8);
        }
    }
    __syncthreads();
}

// ---------------- GRU recurrence ----------------
// NWG_GRU workgroups, each owns 64 columns of h. wh slice (3 x 64 cols x 256 kpair u32
// = 192KB) lives in LDS for the whole kernel (CDNA5: 320KB/WGP). Each (gate, wg) slice
// is a flat contiguous 16384-dword tile in global -> one TDM descriptor per gate.
// h exchanged via L2 each step: bf16 mirror async-b128 staged into LDS [b][kpair].
__global__ __launch_bounds__(256) void gru_kernel(const float* __restrict__ gi,
                                                  const unsigned int* __restrict__ wh_pk,
                                                  const float* __restrict__ bh_n,
                                                  float* __restrict__ hbuf,
                                                  unsigned short* __restrict__ hbf,
                                                  unsigned short* __restrict__ hs,
                                                  float* __restrict__ carry_out,
                                                  int* __restrict__ bar) {
    __shared__ __align__(16) unsigned int lds_wh[3 * 64 * 256];  // [gate][j][kpair] 192 KB
    __shared__ __align__(16) unsigned int lds_h [32 * 256];      // [b][kpair]        32 KB
    const int tid  = threadIdx.x;
    const int lane = tid & 31;
    const int wave = tid >> 5;
    const int wg   = blockIdx.x;
    const int mtile = wave >> 2;       // 0..1  (batch tile)
    const int ntile = wave & 3;        // 0..3  (16-col tile within this WG's 64 cols)
    const int nl   = lane & 15;
    const int half = lane >> 4;
    int* bar_cnt = bar;
    int* bar_gen = bar + 1;

    // ---- one-shot weight preload into LDS: 3 contiguous 64KB tiles ----
#if HAVE_TDM
    if (wave == 0) {
        #pragma unroll
        for (int gate = 0; gate < 3; ++gate) {
            // D# per ISA 08_async_tensor §8.3/8.4: flat 16384-dword tile
            unsigned lds_addr = (unsigned)(uintptr_t)&lds_wh[gate * 16384];
            unsigned long long ga = (unsigned long long)(uintptr_t)
                (wh_pk + (size_t)(gate * H_ + wg * 64) * (H_ / 2));
            u32x4 g0 = { 1u,                                     // count=1, gather off
                         lds_addr,                                // lds_addr
                         (unsigned)(ga & 0xFFFFFFFFu),            // global_addr[31:0]
                         (unsigned)((ga >> 32) & 0x1FFFFFFu) | (2u << 30) }; // [56:32]|type=2
            i32x8 g1 = { (int)(2u << 16),          // data_size=4B; no pad/iter/multicast
                         (int)(16384u << 16),      // tensor_dim0[15:0] @ bits 63:48
                         0,                        // dim0[31:16]=0 | tensor_dim1 lo16 = 0
                         (int)(16384u << 16),      // dim1 hi=0 | tile_dim0=16384
                         (int)(1u),                // tile_dim1=1 | tile_dim2=0
                         (int)(16384u),            // tensor_dim0_stride[31:0]
                         0, 0 };                   // stride hi, dim1_stride
            i32x4 gz = { 0, 0, 0, 0 };
#if defined(__AMDGCN__) && (__clang_major__ >= 23)
            i32x8 z8 = { 0, 0, 0, 0, 0, 0, 0, 0 };
            __builtin_amdgcn_tensor_load_to_lds(g0, g1, gz, gz, z8, 0);
#else
            __builtin_amdgcn_tensor_load_to_lds(g0, g1, gz, gz, 0);
#endif
        }
        __builtin_amdgcn_s_wait_tensorcnt(0);
    }
#else
    for (int idx = tid; idx < 3 * 64 * 256; idx += 256) {
        int gate = idx >> 14, rem = idx & 16383;   // rem = j*256 + kp
        lds_wh[idx] = wh_pk[(size_t)(gate * H_ + wg * 64) * (H_ / 2) + rem];
    }
#endif
    __syncthreads();

    for (int t = 0; t < T_; ++t) {
        const float*          hprev   = hbuf + (t & 1) * (B_ * H_);
        float*                hnext   = hbuf + ((t + 1) & 1) * (B_ * H_);
        const unsigned short* hbfprev = hbf + (t & 1) * (B_ * H_);
        unsigned short*       hbfnext = hbf + ((t + 1) & 1) * (B_ * H_);

        // stage h_prev (bf16 mirror) -> LDS lds_h[b][kpair]; rows contiguous both sides
#if HAVE_ASYNC
        #pragma unroll
        for (int it = 0; it < 8; ++it) {
            int idx = tid + it * 256;          // 0..2047: b = idx/64, quad = idx%64
            int b = idx >> 6, qd = idx & 63;
            async_b128(hbfprev + b * H_ + qd * 8, &lds_h[b * 256 + qd * 4]);
        }
        __builtin_amdgcn_s_wait_asynccnt(0);
#else
        for (int p = tid; p < (B_ * H_ / 2); p += 256) {
            int b = p >> 8, kp = p & 255;
            float2 hv = *(const float2*)(hprev + b * H_ + kp * 2);
            lds_h[b * 256 + kp] = pack_bf16x2(hv.x, hv.y);
        }
#endif
        __syncthreads();

        v8f ar = {}, az = {}, an = {};
        const int m  = mtile * 16 + nl;
        const int cj = ntile * 16 + nl;
        for (int ks = 0; ks < H_ / 32; ++ks) {
            const int base = ks * 16 + half * 4;
            Frag af, br, bz, bn;
            af.q[0] = *(const u32x4*)&lds_h[m * 256 + base];
            af.q[1] = *(const u32x4*)&lds_h[m * 256 + base + 8];
            br.q[0] = *(const u32x4*)&lds_wh[(0 * 64 + cj) * 256 + base];
            br.q[1] = *(const u32x4*)&lds_wh[(0 * 64 + cj) * 256 + base + 8];
            bz.q[0] = *(const u32x4*)&lds_wh[(1 * 64 + cj) * 256 + base];
            bz.q[1] = *(const u32x4*)&lds_wh[(1 * 64 + cj) * 256 + base + 8];
            bn.q[0] = *(const u32x4*)&lds_wh[(2 * 64 + cj) * 256 + base];
            bn.q[1] = *(const u32x4*)&lds_wh[(2 * 64 + cj) * 256 + base + 8];
            ar = __builtin_amdgcn_wmma_f32_16x16x32_bf16(false, af.v, false, br.v, (short)0, ar, false, false);
            az = __builtin_amdgcn_wmma_f32_16x16x32_bf16(false, af.v, false, bz.v, (short)0, az, false, false);
            an = __builtin_amdgcn_wmma_f32_16x16x32_bf16(false, af.v, false, bn.v, (short)0, an, false, false);
        }

        // elementwise gate math + h update for this wave's 16x16 C-tile
        {
            const int col = wg * 64 + cj;
            const float bhn = bh_n[col];
            const float* girow = gi + (size_t)t * B_ * G3H;
            #pragma unroll
            for (int r = 0; r < 8; ++r) {
                int b = mtile * 16 + r + half * 8;
                const float* g = girow + b * G3H;
                float rv = sigmoidf_(ar[r] + g[col]);
                float zv = sigmoidf_(az[r] + g[H_ + col]);
                float nv = tanhf(g[2 * H_ + col] + rv * (an[r] + bhn));
                float hp = hprev[b * H_ + col];
                float hn = (1.0f - zv) * nv + zv * hp;
                hnext[b * H_ + col] = hn;
                unsigned short hb = f32_to_bf16_bits(hn);
                hbfnext[b * H_ + col] = hb;
                hs[((size_t)b * T_ + t) * H_ + col] = hb;
                if (t == T_ - 1) carry_out[b * H_ + col] = hn;
            }
        }
        grid_barrier(bar_cnt, bar_gen, NWG_GRU);
    }
}

// ---------------- launcher ----------------
extern "C" void kernel_launch(void* const* d_in, const int* in_sizes, int n_in,
                              void* d_out, int out_size, void* d_ws, size_t ws_size,
                              hipStream_t stream) {
    const int*   x     = (const int*)  d_in[0];
    const float* embed = (const float*)d_in[1];
    const float* wi_r  = (const float*)d_in[2];
    const float* bi_r  = (const float*)d_in[3];
    const float* wi_z  = (const float*)d_in[4];
    const float* bi_z  = (const float*)d_in[5];
    const float* wi_n  = (const float*)d_in[6];
    const float* bi_n  = (const float*)d_in[7];
    const float* wh_r  = (const float*)d_in[8];
    const float* wh_z  = (const float*)d_in[9];
    const float* wh_n  = (const float*)d_in[10];
    const float* bh_n  = (const float*)d_in[11];
    const float* wd    = (const float*)d_in[12];
    const float* bd    = (const float*)d_in[13];

    float* out = (float*)d_out;
    char*  ws  = (char*)d_ws;

    float*          hbuf     = (float*)(ws + WS_HBUF);
    unsigned short* hbf      = (unsigned short*)(ws + WS_HBF);
    int*            bar      = (int*)(ws + WS_BAR);
    float*          bias_cat = (float*)(ws + WS_BIAS);
    unsigned int*   wi_pk    = (unsigned int*)(ws + WS_WI_PK);
    unsigned int*   wh_pk    = (unsigned int*)(ws + WS_WH_PK);
    unsigned int*   wd_pk    = (unsigned int*)(ws + WS_WD_PK);
    unsigned short* emba     = (unsigned short*)(ws + WS_EMBA);
    unsigned short* hs       = (unsigned short*)(ws + WS_HS);
    float*          gi       = (float*)(ws + WS_GI);

    // init + weight packing (bf16 k-major pair layouts)
    init_kernel<<<128, 256, 0, stream>>>(hbuf, (unsigned int*)hbf, bar,
                                         bi_r, bi_z, bi_n, bias_cat);
    pack3_kernel<<<(E_/2)*G3H/256, 256, 0, stream>>>(wi_r, wi_z, wi_n, wi_pk, E_);
    pack3_kernel<<<(H_/2)*G3H/256, 256, 0, stream>>>(wh_r, wh_z, wh_n, wh_pk, H_);
    pack1_kernel<<<(H_/2)*V_/256, 256, 0, stream>>>(wd, wd_pk, H_, V_);

    // embedding gather (writes d_out emb section + bf16 GEMM operand)
    embed_kernel<<<(size_t)BT_*E_/256, 256, 0, stream>>>(x, embed, out + OUT_EMB, emba);

    // hoisted input projection: gi[T*B, 3H] = emb @ [wi_r|wi_z|wi_n] + [bi]
    gemm_kernel<<<dim3(G3H/128, BT_/128), 256, 0, stream>>>(emba, wi_pk, bias_cat, gi, G3H, E_);

    // sequential GRU with LDS-resident weights
    gru_kernel<<<NWG_GRU, 256, 0, stream>>>(gi, wh_pk, bh_n, hbuf, hbf, hs,
                                            out + OUT_CARRY, bar);

    // output projection: logits[B*T, V] = hs @ wd + bd
    gemm_kernel<<<dim3(V_/128, BT_/128), 256, 0, stream>>>(hs, wd_pk, bd, out, V_, H_);
}
